// IC50Predictor_81853486727684
// MI455X (gfx1250) — compile-verified, matrix operand order checked by reading, not run on hardware
//
#include <hip/hip_runtime.h>
#include <hip/hip_bf16.h>
#include <stddef.h>

// ---------------- problem sizes ----------------
#define HHID 256
#define DMD  32
#define FFD  64
#define BBAT 2048
#define SSEQ 192
#define NNOD 81920
#define EEDG 245760
#define TTOK (BBAT*SSEQ)   // 393216

typedef __bf16 bf16;
typedef __attribute__((ext_vector_type(16))) __bf16 v16bf;
typedef __attribute__((ext_vector_type(8)))  float  v8f;

// ---------------- device helpers ----------------
__device__ __forceinline__ float warp_sum(float v) {
#pragma unroll
  for (int o = 16; o > 0; o >>= 1) v += __shfl_xor(v, o, 32);
  return v;
}
// monotonic float<->uint encoding for atomicMax-based segment max
__device__ __forceinline__ unsigned fenc(float f) {
  unsigned u = __float_as_uint(f);
  return (u & 0x80000000u) ? ~u : (u | 0x80000000u);
}
__device__ __forceinline__ float fdec(unsigned u) {
  u = (u & 0x80000000u) ? (u & 0x7FFFFFFFu) : ~u;
  return __uint_as_float(u);
}
__device__ __forceinline__ float lrelu(float x) { return x > 0.f ? x : 0.01f * x; }
__device__ __forceinline__ float eluf(float x)  { return x > 0.f ? x : (__expf(x) - 1.f); }
__device__ __forceinline__ float sigm(float x)  { return 1.f / (1.f + __expf(-x)); }

// ---------------- generic bf16 WMMA GEMM ----------------
// C[M,Nn] (f32) = act( A[M,K](bf16,row-major,lda) * W[Nn,K](bf16,row-major,ldw)^T + bias )
// optional bf16 mirror of C in Cb.  One wave computes one 16x16 C tile.
// act: 0=none 1=relu 2=leaky_relu(0.01) 3=elu
__global__ __launch_bounds__(128) void k_gemm(
    const bf16* __restrict__ A, int lda, const bf16* __restrict__ W, int ldw,
    const float* __restrict__ bias, float* __restrict__ C, int ldc,
    int M, int Nn, int K, int act, bf16* __restrict__ Cb)
{
  int tile = blockIdx.x * 4 + (threadIdx.x >> 5);
  int ntn  = Nn >> 4;
  int ntot = (M >> 4) * ntn;
  if (tile >= ntot) return;                     // uniform per wave -> EXEC all ones
  int tm = (tile / ntn) << 4;
  int tn = (tile % ntn) << 4;
  int lane = threadIdx.x & 31;
  int r  = lane & 15;
  int hh = lane >> 4;                           // half: K-sub-block 0/1
  const bf16* arow = A + (size_t)(tm + r) * lda + hh * 16;
  const bf16* wrow = W + (size_t)(tn + r) * ldw + hh * 16;
  v8f acc = {};
  for (int k0 = 0; k0 < K; k0 += 32) {
    __builtin_prefetch((const void*)(arow + k0 + 128), 0, 3);  // global_prefetch_b8 (near)
    v16bf av, bv;
#pragma unroll
    for (int i = 0; i < 16; i++) { av[i] = arow[k0 + i]; bv[i] = wrow[k0 + i]; }
    acc = __builtin_amdgcn_wmma_f32_16x16x32_bf16(false, av, false, bv,
                                                  (short)0, acc, false, false);
  }
  // C/D layout: lanes 0-15 -> N=lane, M=v ; lanes 16-31 -> N=lane-16, M=v+8
  int n = tn + r;
  float bval = bias ? bias[n] : 0.f;
#pragma unroll
  for (int v = 0; v < 8; v++) {
    int m = tm + v + hh * 8;
    float x = acc[v] + bval;
    if (act == 1)      x = fmaxf(x, 0.f);
    else if (act == 2) x = lrelu(x);
    else if (act == 3) x = eluf(x);
    C[(size_t)m * ldc + n] = x;
    if (Cb) Cb[(size_t)m * ldc + n] = (bf16)x;
  }
}

// ---------------- elementwise / conversion kernels ----------------
__global__ void k_cvt_bf16(const float* __restrict__ s, bf16* __restrict__ d, size_t n) {
  size_t i = (size_t)blockIdx.x * blockDim.x + threadIdx.x;
  if (i < n) d[i] = (bf16)s[i];
}
__global__ void k_cvt_bf16_strided(const float* __restrict__ s, bf16* __restrict__ d,
                                   int rows, int cols, int ld) {
  size_t i = (size_t)blockIdx.x * blockDim.x + threadIdx.x;
  if (i >= (size_t)rows * cols) return;
  int rr = (int)(i / cols), cc = (int)(i % cols);
  d[i] = (bf16)s[(size_t)rr * ld + cc];
}
__global__ void k_embed(const int* __restrict__ seq, const float* __restrict__ emb,
                        float* __restrict__ x, bf16* __restrict__ xb) {
  size_t i = (size_t)blockIdx.x * blockDim.x + threadIdx.x;
  if (i >= (size_t)TTOK * DMD) return;
  size_t t = i >> 5; int d = (int)(i & 31);
  float v = emb[seq[t] * DMD + d];
  x[i] = v; xb[i] = (bf16)v;
}
// ---------------- protein attention (2 heads, hd=16, S=192) ----------------
// K/V tiles staged into LDS with CDNA5 async copies (ASYNCcnt) instead of
// VGPR round-trips: each lane DMAs its 64B K row and V row via 4x b128 ops,
// then s_wait_asynccnt 0 + workgroup barrier.
__global__ __launch_bounds__(SSEQ) void k_attn(
    const float* __restrict__ q, const float* __restrict__ k,
    const float* __restrict__ v, float* __restrict__ o, bf16* __restrict__ ob)
{
  int bh = blockIdx.x; int b = bh >> 1; int hh = bh & 1;
  __shared__ float ks[SSEQ][16];
  __shared__ float vs[SSEQ][16];
  int t = threadIdx.x;
  size_t base = ((size_t)b * SSEQ + t) * DMD + hh * 16;
  unsigned lk = (unsigned)(size_t)&ks[t][0];          // LDS byte offset (addr[31:0])
  unsigned lv = (unsigned)(size_t)&vs[t][0];
  unsigned long long gk = (unsigned long long)(const void*)(k + base);
  unsigned long long gv = (unsigned long long)(const void*)(v + base);
  // instruction offset applies to both LDS and global sides (ISA 10.7 / 15.18.3)
  asm volatile(
      "global_load_async_to_lds_b128 %0, %2, off\n\t"
      "global_load_async_to_lds_b128 %0, %2, off offset:16\n\t"
      "global_load_async_to_lds_b128 %0, %2, off offset:32\n\t"
      "global_load_async_to_lds_b128 %0, %2, off offset:48\n\t"
      "global_load_async_to_lds_b128 %1, %3, off\n\t"
      "global_load_async_to_lds_b128 %1, %3, off offset:16\n\t"
      "global_load_async_to_lds_b128 %1, %3, off offset:32\n\t"
      "global_load_async_to_lds_b128 %1, %3, off offset:48"
      :: "v"(lk), "v"(lv), "v"(gk), "v"(gv) : "memory");
  asm volatile("s_wait_asynccnt 0" ::: "memory");
  __syncthreads();
  float qr[16];
#pragma unroll
  for (int d = 0; d < 16; d++) qr[d] = q[base + d];
  float mx = -1e30f;
  for (int j = 0; j < SSEQ; j++) {
    float s = 0.f;
#pragma unroll
    for (int d = 0; d < 16; d++) s += qr[d] * ks[j][d];
    mx = fmaxf(mx, s * 0.25f);
  }
  float sum = 0.f, acc[16] = {};
  for (int j = 0; j < SSEQ; j++) {
    float s = 0.f;
#pragma unroll
    for (int d = 0; d < 16; d++) s += qr[d] * ks[j][d];
    float e = __expf(s * 0.25f - mx);
    sum += e;
#pragma unroll
    for (int d = 0; d < 16; d++) acc[d] += e * vs[j][d];
  }
  float inv = 1.f / sum;
#pragma unroll
  for (int d = 0; d < 16; d++) {
    float val = acc[d] * inv;
    o[base + d] = val; ob[base + d] = (bf16)val;
  }
}
// residual add + layernorm over 32 dims (one token per wave32)
__global__ void k_ln(const float* __restrict__ a, const float* __restrict__ r,
                     const float* __restrict__ g, const float* __restrict__ bb,
                     float* __restrict__ out, bf16* __restrict__ outb) {
  size_t i = (size_t)blockIdx.x * blockDim.x + threadIdx.x;
  if (i >= (size_t)TTOK * DMD) return;
  float x = a[i] + r[i];
  float m = warp_sum(x) * (1.f / 32.f);
  float dlt = x - m;
  float var = warp_sum(dlt * dlt) * (1.f / 32.f);
  float y = dlt * rsqrtf(var + 1e-5f) * g[i & 31] + bb[i & 31];
  out[i] = y; if (outb) outb[i] = (bf16)y;
}
__global__ void k_seqmean(const float* __restrict__ x, float* __restrict__ out) {
  size_t i = (size_t)blockIdx.x * blockDim.x + threadIdx.x;
  if (i >= (size_t)BBAT * DMD) return;
  size_t b = i >> 5; int d = (int)(i & 31);
  float s = 0.f;
  for (int t = 0; t < SSEQ; t++) s += x[((size_t)b * SSEQ + t) * DMD + d];
  out[i] = s * (1.f / (float)SSEQ);
}
// ---------------- ligand kernels ----------------
__global__ void k_lig_lin1(const float* __restrict__ nx, const float* __restrict__ w,
                           const float* __restrict__ b, float* __restrict__ x,
                           bf16* __restrict__ xb) {
  size_t i = (size_t)blockIdx.x * blockDim.x + threadIdx.x;
  if (i >= (size_t)NNOD * HHID) return;
  size_t n = i >> 8; int o = (int)(i & 255);
  float s = b[o];
#pragma unroll
  for (int k = 0; k < 3; k++) s += nx[n * 3 + k] * w[o * 3 + k];
  s = lrelu(s);
  x[i] = s; xb[i] = (bf16)s;
}
// one wave per row: out[row] = dot(x[row,:256], vec)
__global__ void k_rowdot(const float* __restrict__ x, const float* __restrict__ vec,
                         float* __restrict__ out, int rows) {
  int row = blockIdx.x * (blockDim.x >> 5) + (threadIdx.x >> 5);
  if (row >= rows) return;
  int lane = threadIdx.x & 31;
  float s = 0.f;
  for (int c = lane; c < HHID; c += 32) s += x[(size_t)row * HHID + c] * vec[c];
  s = warp_sum(s);
  if (lane == 0) out[row] = s;
}
// GATEConv edge logit: lrelu( dot(lrelu(xw[src]+ea*u), att_l) + xr[dst] )
__global__ void k_gate_logit(const float* __restrict__ xw, const float* __restrict__ l1w,
                             const float* __restrict__ ea, const float* __restrict__ attl,
                             const float* __restrict__ xr, const int* __restrict__ src,
                             const int* __restrict__ dst, float* __restrict__ out) {
  int e = blockIdx.x * (blockDim.x >> 5) + (threadIdx.x >> 5);
  if (e >= EEDG) return;
  int lane = threadIdx.x & 31;
  int si = src[e];
  float eav = ea[e];
  float s = 0.f;
  for (int c = lane; c < HHID; c += 32) {
    float u = l1w[(size_t)c * 257 + 256];          // last column of [256,257]
    float t = lrelu(xw[(size_t)si * HHID + c] + eav * u);
    s += t * attl[c];
  }
  s = warp_sum(s);
  if (lane == 0) out[e] = lrelu(s + xr[dst[e]]);
}
__global__ void k_gat_logit(const float* __restrict__ as_, const float* __restrict__ ad_,
                            const int* __restrict__ src, const int* __restrict__ dst,
                            float* __restrict__ out, int n) {
  int i = blockIdx.x * blockDim.x + threadIdx.x;
  if (i < n) out[i] = lrelu(as_[src[i]] + ad_[dst[i]]);
}
__global__ void k_logit_node(const float* __restrict__ as_, const float* __restrict__ ad_,
                             const int* __restrict__ seg, float* __restrict__ out, int n) {
  int i = blockIdx.x * blockDim.x + threadIdx.x;
  if (i < n) out[i] = lrelu(as_[i] + ad_[seg[i]]);
}
__global__ void k_segmax(const float* __restrict__ a, const int* __restrict__ seg,
                         unsigned* __restrict__ mx, int n) {
  int i = blockIdx.x * blockDim.x + threadIdx.x;
  if (i < n) atomicMax(&mx[seg[i]], fenc(a[i]));
}
__global__ void k_segexp(const float* __restrict__ a, const int* __restrict__ seg,
                         const unsigned* __restrict__ mx, float* __restrict__ w,
                         float* __restrict__ sum, int n) {
  int i = blockIdx.x * blockDim.x + threadIdx.x;
  if (i < n) {
    float e = __expf(a[i] - fdec(mx[seg[i]]));
    w[i] = e;
    atomicAdd(&sum[seg[i]], e);
  }
}
// out[seg[e], c] += vals[src[e], c] * w[e] / (sum[seg[e]]+eps);  src==null => src[e]=e
__global__ void k_scatter(const float* __restrict__ vals, const int* __restrict__ src,
                          const int* __restrict__ seg, const float* __restrict__ w,
                          const float* __restrict__ sum, float* __restrict__ out,
                          size_t n_items) {
  size_t i = (size_t)blockIdx.x * blockDim.x + threadIdx.x;
  if (i >= n_items * HHID) return;
  size_t e = i >> 8; int c = (int)(i & 255);
  int sr = src ? src[e] : (int)e;
  int sg = seg[e];
  float alpha = w[e] / (sum[sg] + 1e-16f);
  atomicAdd(&out[(size_t)sg * HHID + c], vals[(size_t)sr * HHID + c] * alpha);
}
__global__ void k_segsum256(const float* __restrict__ x, const int* __restrict__ seg,
                            float* __restrict__ out, size_t rows) {
  size_t i = (size_t)blockIdx.x * blockDim.x + threadIdx.x;
  if (i >= rows * HHID) return;
  size_t n = i >> 8; int c = (int)(i & 255);
  atomicAdd(&out[(size_t)seg[n] * HHID + c], x[i]);
}
__global__ void k_bias_elu(const float* __restrict__ in, const float* __restrict__ b,
                           float* __restrict__ out, bf16* __restrict__ outb, size_t n) {
  size_t i = (size_t)blockIdx.x * blockDim.x + threadIdx.x;
  if (i >= n) return;
  float x = eluf(in[i] + b[i & 255]);
  out[i] = x; outb[i] = (bf16)x;
}
__global__ void k_relu_bf(const float* __restrict__ in, float* __restrict__ out,
                          bf16* __restrict__ outb, size_t n) {
  size_t i = (size_t)blockIdx.x * blockDim.x + threadIdx.x;
  if (i >= n) return;
  float x = fmaxf(in[i], 0.f);
  out[i] = x; outb[i] = (bf16)x;
}
// GRUCell gates + relu; x updated in place (per-element read-then-write is safe)
__global__ void k_gru(const float* __restrict__ gi, const float* __restrict__ gh,
                      float* __restrict__ x, bf16* __restrict__ xb, size_t rows) {
  size_t i = (size_t)blockIdx.x * blockDim.x + threadIdx.x;
  if (i >= rows * HHID) return;
  size_t n = i >> 8; int c = (int)(i & 255);
  size_t b3 = n * 3 * HHID;
  float r = sigm(gi[b3 + c] + gh[b3 + c]);
  float z = sigm(gi[b3 + HHID + c] + gh[b3 + HHID + c]);
  float nn = tanhf(gi[b3 + 2 * HHID + c] + r * gh[b3 + 2 * HHID + c]);
  float h = fmaxf((1.f - z) * nn + z * x[i], 0.f);
  x[i] = h; xb[i] = (bf16)h;
}
// ---------------- final head ----------------
__global__ void k_head(const float* __restrict__ prot, const float* __restrict__ lig,
                       const float* __restrict__ w1, const float* __restrict__ b1,
                       const float* __restrict__ w2, const float* __restrict__ b2,
                       float* __restrict__ y) {
  int b = blockIdx.x * blockDim.x + threadIdx.x;
  if (b >= BBAT) return;
  float z[64];
#pragma unroll
  for (int j = 0; j < 32; j++) { z[j] = prot[b * 32 + j]; z[32 + j] = lig[b * 32 + j]; }
  float s2 = b2[0];
  for (int o = 0; o < 32; o++) {
    float s = b1[o];
#pragma unroll
    for (int k = 0; k < 64; k++) s += z[k] * w1[o * 64 + k];
    s2 += fmaxf(s, 0.f) * w2[o];
  }
  y[b] = s2;
}

// ---------------- host orchestration ----------------
static inline void gemm(hipStream_t st, const bf16* A, int lda, const bf16* W, int ldw,
                        const float* bias, float* C, int ldc, int M, int Nn, int K,
                        int act, bf16* Cb) {
  int tiles = (M / 16) * (Nn / 16);
  k_gemm<<<(tiles + 3) / 4, 128, 0, st>>>(A, lda, W, ldw, bias, C, ldc, M, Nn, K, act, Cb);
}
static inline dim3 gr(size_t n, int t) { return dim3((unsigned)((n + t - 1) / t)); }

extern "C" void kernel_launch(void* const* d_in, const int* in_sizes, int n_in,
                              void* d_out, int out_size, void* d_ws, size_t ws_size,
                              hipStream_t stream) {
  (void)in_sizes; (void)n_in; (void)out_size; (void)ws_size;
  // ---- inputs (params flattened depth-first in dict insertion order) ----
  const int*   seq   = (const int*)d_in[0];
  const float* nodex = (const float*)d_in[1];
  const int*   esrc  = (const int*)d_in[2];
  const int*   edst  = esrc + EEDG;
  const float* eattr = (const float*)d_in[3];
  const int*   batch = (const int*)d_in[4];
  const float* lg_l1w = (const float*)d_in[5];  const float* lg_l1b = (const float*)d_in[6];
  const float* gt_l1w = (const float*)d_in[7];  const float* gt_l2w = (const float*)d_in[8];
  const float* gt_al  = (const float*)d_in[9];  const float* gt_ar  = (const float*)d_in[10];
  const float* gt_b   = (const float*)d_in[11];
  struct GruP { const float *wi, *wh, *bi, *bh; };
  struct GatP { const float *w, *asr, *adt, *b; };
  GruP gru0 = { (const float*)d_in[12], (const float*)d_in[13],
                (const float*)d_in[14], (const float*)d_in[15] };
  GatP ag[3]; GruP ar[3];
  for (int i = 0; i < 3; i++) {
    int o = 16 + i * 8;
    ag[i] = { (const float*)d_in[o],     (const float*)d_in[o + 1],
              (const float*)d_in[o + 2], (const float*)d_in[o + 3] };
    ar[i] = { (const float*)d_in[o + 4], (const float*)d_in[o + 5],
              (const float*)d_in[o + 6], (const float*)d_in[o + 7] };
  }
  GatP mg = { (const float*)d_in[40], (const float*)d_in[41],
              (const float*)d_in[42], (const float*)d_in[43] };
  GruP mr = { (const float*)d_in[44], (const float*)d_in[45],
              (const float*)d_in[46], (const float*)d_in[47] };
  const float* lg_l2w = (const float*)d_in[48]; const float* lg_l2b = (const float*)d_in[49];
  const float* p_emb  = (const float*)d_in[50];
  const float* wq = (const float*)d_in[51]; const float* bq = (const float*)d_in[52];
  const float* wk = (const float*)d_in[53]; const float* bk = (const float*)d_in[54];
  const float* wv = (const float*)d_in[55]; const float* bv = (const float*)d_in[56];
  const float* wo = (const float*)d_in[57]; const float* bo = (const float*)d_in[58];
  const float* ln1g = (const float*)d_in[59]; const float* ln1b = (const float*)d_in[60];
  const float* ln2g = (const float*)d_in[61]; const float* ln2b = (const float*)d_in[62];
  const float* f1w = (const float*)d_in[63]; const float* f1b = (const float*)d_in[64];
  const float* f2w = (const float*)d_in[65]; const float* f2b = (const float*)d_in[66];
  const float* fcw1 = (const float*)d_in[67]; const float* fcb1 = (const float*)d_in[68];
  const float* fcw2 = (const float*)d_in[69]; const float* fcb2 = (const float*)d_in[70];

  // ---- workspace allocator ----
  char* wsb = (char*)d_ws; size_t off = 0;
  auto alloc = [&](size_t bytes) -> void* {
    void* p = wsb + off; off = (off + bytes + 255) & ~(size_t)255; return p;
  };
  // bf16 weight mirrors
  bf16* wb_g1  = (bf16*)alloc((size_t)256 * 256 * 2);
  bf16* wb_g2  = (bf16*)alloc((size_t)256 * 256 * 2);
  bf16* wb_0wi = (bf16*)alloc((size_t)768 * 256 * 2);
  bf16* wb_0wh = (bf16*)alloc((size_t)768 * 256 * 2);
  bf16 *wb_aw[3], *wb_awi[3], *wb_awh[3];
  for (int i = 0; i < 3; i++) {
    wb_aw[i]  = (bf16*)alloc((size_t)256 * 256 * 2);
    wb_awi[i] = (bf16*)alloc((size_t)768 * 256 * 2);
    wb_awh[i] = (bf16*)alloc((size_t)768 * 256 * 2);
  }
  bf16* wb_mw  = (bf16*)alloc((size_t)256 * 256 * 2);
  bf16* wb_mwi = (bf16*)alloc((size_t)768 * 256 * 2);
  bf16* wb_mwh = (bf16*)alloc((size_t)768 * 256 * 2);
  bf16* wb_q = (bf16*)alloc(1024 * 2); bf16* wb_k = (bf16*)alloc(1024 * 2);
  bf16* wb_v = (bf16*)alloc(1024 * 2); bf16* wb_o = (bf16*)alloc(1024 * 2);
  bf16* wb_f1 = (bf16*)alloc(2048 * 2); bf16* wb_f2 = (bf16*)alloc(2048 * 2);
  bf16* wb_l2 = (bf16*)alloc(8192 * 2);
  // protein activations
  float* p_x  = (float*)alloc((size_t)TTOK * 32 * 4);  bf16* p_xb  = (bf16*)alloc((size_t)TTOK * 32 * 2);
  float* p_q  = (float*)alloc((size_t)TTOK * 32 * 4);
  float* p_k  = (float*)alloc((size_t)TTOK * 32 * 4);
  float* p_v  = (float*)alloc((size_t)TTOK * 32 * 4);
  float* p_at = (float*)alloc((size_t)TTOK * 32 * 4);  bf16* p_atb = (bf16*)alloc((size_t)TTOK * 32 * 2);
  float* p_o  = (float*)alloc((size_t)TTOK * 32 * 4);
  float* p_l1 = (float*)alloc((size_t)TTOK * 32 * 4);  bf16* p_l1b = (bf16*)alloc((size_t)TTOK * 32 * 2);
  float* p_f1 = (float*)alloc((size_t)TTOK * 64 * 4);  bf16* p_f1b = (bf16*)alloc((size_t)TTOK * 64 * 2);
  float* p_f2 = (float*)alloc((size_t)TTOK * 32 * 4);
  float* p_l2 = (float*)alloc((size_t)TTOK * 32 * 4);
  float* protv = (float*)alloc((size_t)BBAT * 32 * 4);
  // ligand activations
  float* l_x   = (float*)alloc((size_t)NNOD * HHID * 4);  bf16* l_xb = (bf16*)alloc((size_t)NNOD * HHID * 2);
  float* l_h   = (float*)alloc((size_t)NNOD * HHID * 4);  bf16* l_hb = (bf16*)alloc((size_t)NNOD * HHID * 2);
  float* l_xs  = (float*)alloc((size_t)NNOD * HHID * 4);
  float* l_xl2 = (float*)alloc((size_t)NNOD * HHID * 4);  // reused as xs_mol
  float* l_agg = (float*)alloc((size_t)NNOD * HHID * 4);
  float* l_gi  = (float*)alloc((size_t)NNOD * 3 * HHID * 4);
  float* l_gh  = (float*)alloc((size_t)NNOD * 3 * HHID * 4);
  float* e_a   = (float*)alloc((size_t)EEDG * 4);         // also node logits (N<=E)
  float* e_w   = (float*)alloc((size_t)EEDG * 4);
  unsigned* n_max = (unsigned*)alloc((size_t)NNOD * 4);
  float* n_sum = (float*)alloc((size_t)NNOD * 4);
  float* n_xr  = (float*)alloc((size_t)NNOD * 4);
  float* n_as  = (float*)alloc((size_t)NNOD * 4);
  float* n_ad  = (float*)alloc((size_t)NNOD * 4);
  float* outB  = (float*)alloc((size_t)BBAT * HHID * 4); bf16* outBb = (bf16*)alloc((size_t)BBAT * HHID * 2);
  float* xdB   = (float*)alloc((size_t)BBAT * HHID * 4);
  float* adB   = (float*)alloc((size_t)BBAT * 4);
  unsigned* B_max = (unsigned*)alloc((size_t)BBAT * 4);
  float* B_sum = (float*)alloc((size_t)BBAT * 4);
  float* aggB  = (float*)alloc((size_t)BBAT * HHID * 4);
  float* hB    = (float*)alloc((size_t)BBAT * HHID * 4); bf16* hBb = (bf16*)alloc((size_t)BBAT * HHID * 2);
  float* giB   = (float*)alloc((size_t)BBAT * 3 * HHID * 4);
  float* ghB   = (float*)alloc((size_t)BBAT * 3 * HHID * 4);
  float* ligv  = (float*)alloc((size_t)BBAT * 32 * 4);

  // ---- weight conversions (every call; graph-capture safe) ----
  auto cvt = [&](const float* s, bf16* d, size_t n) {
    k_cvt_bf16<<<gr(n, 256), 256, 0, stream>>>(s, d, n);
  };
  k_cvt_bf16_strided<<<gr((size_t)256 * 256, 256), 256, 0, stream>>>(gt_l1w, wb_g1, 256, 256, 257);
  cvt(gt_l2w, wb_g2, 65536);
  cvt(gru0.wi, wb_0wi, 196608); cvt(gru0.wh, wb_0wh, 196608);
  for (int i = 0; i < 3; i++) {
    cvt(ag[i].w, wb_aw[i], 65536);
    cvt(ar[i].wi, wb_awi[i], 196608); cvt(ar[i].wh, wb_awh[i], 196608);
  }
  cvt(mg.w, wb_mw, 65536); cvt(mr.wi, wb_mwi, 196608); cvt(mr.wh, wb_mwh, 196608);
  cvt(wq, wb_q, 1024); cvt(wk, wb_k, 1024); cvt(wv, wb_v, 1024); cvt(wo, wb_o, 1024);
  cvt(f1w, wb_f1, 2048); cvt(f2w, wb_f2, 2048); cvt(lg_l2w, wb_l2, 8192);

  // ================= protein branch =================
  k_embed<<<gr((size_t)TTOK * 32, 256), 256, 0, stream>>>(seq, p_emb, p_x, p_xb);
  gemm(stream, p_xb, 32, wb_q, 32, bq, p_q, 32, TTOK, 32, 32, 0, nullptr);
  gemm(stream, p_xb, 32, wb_k, 32, bk, p_k, 32, TTOK, 32, 32, 0, nullptr);
  gemm(stream, p_xb, 32, wb_v, 32, bv, p_v, 32, TTOK, 32, 32, 0, nullptr);
  k_attn<<<BBAT * 2, SSEQ, 0, stream>>>(p_q, p_k, p_v, p_at, p_atb);
  gemm(stream, p_atb, 32, wb_o, 32, bo, p_o, 32, TTOK, 32, 32, 0, nullptr);
  k_ln<<<gr((size_t)TTOK * 32, 256), 256, 0, stream>>>(p_x, p_o, ln1g, ln1b, p_l1, p_l1b);
  gemm(stream, p_l1b, 32, wb_f1, 32, f1b, p_f1, 64, TTOK, 64, 32, 1, p_f1b);
  gemm(stream, p_f1b, 64, wb_f2, 64, f2b, p_f2, 32, TTOK, 32, 64, 0, nullptr);
  k_ln<<<gr((size_t)TTOK * 32, 256), 256, 0, stream>>>(p_l1, p_f2, ln2g, ln2b, p_l2, nullptr);
  k_seqmean<<<gr((size_t)BBAT * 32, 256), 256, 0, stream>>>(p_l2, protv);

  // ================= ligand branch =================
  k_lig_lin1<<<gr((size_t)NNOD * HHID, 256), 256, 0, stream>>>(nodex, lg_l1w, lg_l1b, l_x, l_xb);
  // gate conv
  gemm(stream, l_xb, HHID, wb_g1, HHID, nullptr, l_xs, HHID, NNOD, HHID, HHID, 0, nullptr);
  gemm(stream, l_xb, HHID, wb_g2, HHID, nullptr, l_xl2, HHID, NNOD, HHID, HHID, 0, nullptr);
  k_rowdot<<<gr((size_t)NNOD * 32, 256), 256, 0, stream>>>(l_x, gt_ar, n_xr, NNOD);
  k_gate_logit<<<gr((size_t)EEDG * 32, 256), 256, 0, stream>>>(l_xs, gt_l1w, eattr, gt_al,
                                                               n_xr, esrc, edst, e_a);
  auto edge_softmax_scatter = [&](const float* vals, const int* src) {
    hipMemsetAsync(n_max, 0, (size_t)NNOD * 4, stream);
    hipMemsetAsync(n_sum, 0, (size_t)NNOD * 4, stream);
    k_segmax<<<gr(EEDG, 256), 256, 0, stream>>>(e_a, edst, n_max, EEDG);
    k_segexp<<<gr(EEDG, 256), 256, 0, stream>>>(e_a, edst, n_max, e_w, n_sum, EEDG);
    hipMemsetAsync(l_agg, 0, (size_t)NNOD * HHID * 4, stream);
    k_scatter<<<gr((size_t)EEDG * HHID, 256), 256, 0, stream>>>(vals, src, edst, e_w, n_sum,
                                                                l_agg, (size_t)EEDG);
  };
  edge_softmax_scatter(l_xl2, esrc);
  k_bias_elu<<<gr((size_t)NNOD * HHID, 256), 256, 0, stream>>>(l_agg, gt_b, l_h, l_hb,
                                                               (size_t)NNOD * HHID);
  auto gru_step = [&](const bf16* wi, const bf16* wh, const float* bi, const float* bh) {
    gemm(stream, l_hb, HHID, wi, HHID, bi, l_gi, 768, NNOD, 768, HHID, 0, nullptr);
    gemm(stream, l_xb, HHID, wh, HHID, bh, l_gh, 768, NNOD, 768, HHID, 0, nullptr);
    k_gru<<<gr((size_t)NNOD * HHID, 256), 256, 0, stream>>>(l_gi, l_gh, l_x, l_xb, NNOD);
  };
  gru_step(wb_0wi, wb_0wh, gru0.bi, gru0.bh);
  // atom GAT+GRU layers
  for (int i = 0; i < 3; i++) {
    gemm(stream, l_xb, HHID, wb_aw[i], HHID, nullptr, l_xs, HHID, NNOD, HHID, HHID, 0, nullptr);
    k_rowdot<<<gr((size_t)NNOD * 32, 256), 256, 0, stream>>>(l_xs, ag[i].asr, n_as, NNOD);
    k_rowdot<<<gr((size_t)NNOD * 32, 256), 256, 0, stream>>>(l_xs, ag[i].adt, n_ad, NNOD);
    k_gat_logit<<<gr(EEDG, 256), 256, 0, stream>>>(n_as, n_ad, esrc, edst, e_a, EEDG);
    edge_softmax_scatter(l_xs, esrc);
    k_bias_elu<<<gr((size_t)NNOD * HHID, 256), 256, 0, stream>>>(l_agg, ag[i].b, l_h, l_hb,
                                                                 (size_t)NNOD * HHID);
    gru_step(wb_awi[i], wb_awh[i], ar[i].bi, ar[i].bh);
  }
  // readout
  hipMemsetAsync(outB, 0, (size_t)BBAT * HHID * 4, stream);
  k_segsum256<<<gr((size_t)NNOD * HHID, 256), 256, 0, stream>>>(l_x, batch, outB, NNOD);
  k_relu_bf<<<gr((size_t)BBAT * HHID, 256), 256, 0, stream>>>(outB, outB, outBb,
                                                              (size_t)BBAT * HHID);
  // hoisted: xs_mol = x @ mol_w^T and its att_src dot (weights constant over T)
  gemm(stream, l_xb, HHID, wb_mw, HHID, nullptr, l_xl2, HHID, NNOD, HHID, HHID, 0, nullptr);
  k_rowdot<<<gr((size_t)NNOD * 32, 256), 256, 0, stream>>>(l_xl2, mg.asr, n_as, NNOD);
  for (int t = 0; t < 4; t++) {
    gemm(stream, outBb, HHID, wb_mw, HHID, nullptr, xdB, HHID, BBAT, HHID, HHID, 0, nullptr);
    k_rowdot<<<gr((size_t)BBAT * 32, 256), 256, 0, stream>>>(xdB, mg.adt, adB, BBAT);
    k_logit_node<<<gr(NNOD, 256), 256, 0, stream>>>(n_as, adB, batch, e_a, NNOD);
    hipMemsetAsync(B_max, 0, (size_t)BBAT * 4, stream);
    hipMemsetAsync(B_sum, 0, (size_t)BBAT * 4, stream);
    k_segmax<<<gr(NNOD, 256), 256, 0, stream>>>(e_a, batch, B_max, NNOD);
    k_segexp<<<gr(NNOD, 256), 256, 0, stream>>>(e_a, batch, B_max, e_w, B_sum, NNOD);
    hipMemsetAsync(aggB, 0, (size_t)BBAT * HHID * 4, stream);
    k_scatter<<<gr((size_t)NNOD * HHID, 256), 256, 0, stream>>>(l_xl2, nullptr, batch, e_w,
                                                                B_sum, aggB, (size_t)NNOD);
    k_bias_elu<<<gr((size_t)BBAT * HHID, 256), 256, 0, stream>>>(aggB, mg.b, hB, hBb,
                                                                 (size_t)BBAT * HHID);
    gemm(stream, hBb, HHID, wb_mwi, HHID, mr.bi, giB, 768, BBAT, 768, HHID, 0, nullptr);
    gemm(stream, outBb, HHID, wb_mwh, HHID, mr.bh, ghB, 768, BBAT, 768, HHID, 0, nullptr);
    k_gru<<<gr((size_t)BBAT * HHID, 256), 256, 0, stream>>>(giB, ghB, outB, outBb, BBAT);
  }
  gemm(stream, outBb, HHID, wb_l2, HHID, lg_l2b, ligv, 32, BBAT, 32, HHID, 0, nullptr);
  // ================= head =================
  k_head<<<gr(BBAT, 256), 256, 0, stream>>>(protv, ligv, fcw1, fcb1, fcw2, fcb2,
                                            (float*)d_out);
}
